// MultiHeadSelfAttention_24326694765133
// MI455X (gfx1250) — matrix operand order repaired and checked by reading, not compiled, not run on hardware
//
#include <hip/hip_runtime.h>
#include <stdint.h>

// ---------------- CDNA5 WMMA / TDM types ----------------
typedef __attribute__((ext_vector_type(16))) __bf16 v16bf;
typedef __attribute__((ext_vector_type(8)))  float  v8f;
typedef __attribute__((ext_vector_type(4)))  unsigned int u32x4;
typedef __attribute__((ext_vector_type(8)))  int   i32x8;
typedef __attribute__((ext_vector_type(4)))  int   i32x4;

#define EMBED 1024
#define HEADS 16
#define HDIM  64
#define SEQ   2048
#define BATCH 4
#define ROWS  (BATCH * SEQ)   // 8192
#define QKV_N (3 * EMBED)     // 3072

struct FragPair { uint4 lo, hi; };

__device__ __forceinline__ v16bf load_frag(const unsigned short* p0,
                                           const unsigned short* p1) {
    FragPair fp;
    fp.lo = *reinterpret_cast<const uint4*>(p0);
    fp.hi = *reinterpret_cast<const uint4*>(p1);
    return __builtin_bit_cast(v16bf, fp);
}

__device__ __forceinline__ unsigned short f32_to_bf16(float f) {
    union { float f; unsigned u; } x; x.f = f;
    unsigned r = x.u + 0x7FFFu + ((x.u >> 16) & 1u);  // round-to-nearest-even
    return (unsigned short)(r >> 16);
}

__device__ __forceinline__ v8f wmma_bf16(v16bf a, v16bf b, v8f c) {
    return __builtin_amdgcn_wmma_f32_16x16x32_bf16(false, a, false, b,
                                                   (short)0, c, false, false);
}

// ---- Tensor Data Mover: 2D tile (bf16 elements) global -> LDS ----
// D# per CDNA5 ISA ch.8: group0 {count, lds_addr, global_addr, type=2},
// group1 {data_size=1(2B), tensor dims, tile dims, dim0 stride}.
__device__ __forceinline__ void tdm_load_2d(unsigned lds_off, const void* gptr,
                                            unsigned tile_w, unsigned tile_h,
                                            unsigned row_stride) {
    unsigned long long ga = (unsigned long long)(uintptr_t)gptr;
    u32x4 g0;
    g0[0] = 1u;                                           // count=1 (valid), user mode
    g0[1] = lds_off;                                      // lds_addr [63:32]
    g0[2] = (unsigned)(ga & 0xFFFFFFFFu);                 // global_addr [95:64]
    g0[3] = (unsigned)((ga >> 32) & 0x01FFFFFFu)          // global_addr [120:96]
            | (2u << 30);                                 // type=2 ("image")
    i32x8 g1;
    g1[0] = (int)(1u << 16);                              // data_size=1 -> 2 bytes
    g1[1] = (int)((tile_w & 0xFFFFu) << 16);              // tensor_dim0[15:0]
    g1[2] = (int)(((tile_w >> 16) & 0xFFFFu)              // tensor_dim0[31:16]
            | ((tile_h & 0xFFFFu) << 16));                // tensor_dim1[15:0]
    g1[3] = (int)(((tile_h >> 16) & 0xFFFFu)              // tensor_dim1[31:16]
            | ((tile_w & 0xFFFFu) << 16));                // tile_dim0
    g1[4] = (int)(tile_h & 0xFFFFu);                      // tile_dim1 (tile_dim2=0)
    g1[5] = (int)row_stride;                              // tensor_dim0_stride lo32
    g1[6] = 0;                                            // stride hi / dim1_stride lo
    g1[7] = 0;
    i32x4 g2 = {0, 0, 0, 0};
    i32x4 g3 = {0, 0, 0, 0};
#if defined(__clang_major__) && (__clang_major__ >= 23)
    i32x8 g4 = {0, 0, 0, 0, 0, 0, 0, 0};
    __builtin_amdgcn_tensor_load_to_lds(g0, g1, g2, g3, g4, 0);
#else
    __builtin_amdgcn_tensor_load_to_lds(g0, g1, g2, g3, 0);
#endif
}

__device__ __forceinline__ unsigned lds_off_u32(const void* p) {
    return (unsigned)(uintptr_t)p;   // low 32 bits of flat shared address = LDS offset
}

// ---------------- conversion kernels ----------------
__global__ void cvt_f32_to_bf16(const float* __restrict__ src,
                                unsigned short* __restrict__ dst, int n) {
    int i = blockIdx.x * blockDim.x + threadIdx.x;
    int stride = gridDim.x * blockDim.x;
    for (; i < n; i += stride) dst[i] = f32_to_bf16(src[i]);
}

// src: [1024][Nrows] f32  ->  dst: [Nrows][1024] bf16
__global__ void transpose_to_bf16(const float* __restrict__ src,
                                  unsigned short* __restrict__ dst, int Nrows) {
    int total = Nrows * EMBED;
    int i = blockIdx.x * blockDim.x + threadIdx.x;
    int stride = gridDim.x * blockDim.x;
    for (; i < total; i += stride) {
        int nr = i >> 10;
        int k  = i & 1023;
        dst[i] = f32_to_bf16(src[(size_t)k * Nrows + nr]);
    }
}

// ---------------- 128x128 GEMM core: TDM double-buffered, K=1024 ----------------
// a_lds/b_lds: [2][128*32] bf16. Wave 0 drives the Tensor Data Mover; all waves
// consume via WMMA. One barrier pair per 32-deep K-step, DMA overlapped with math.
__device__ __forceinline__ void gemm_core_128x128(
    const unsigned short* __restrict__ Ag,   // base of block rows (m0), ld=1024
    const unsigned short* __restrict__ Bg,   // base of block rows (n0) of B^T, ld=1024
    unsigned short* a_lds, unsigned short* b_lds,
    v8f acc[2][4])
{
    const int tid  = threadIdx.x;
    const int lane = tid & 31;
    const int wv   = tid >> 5;
    const int wm   = wv >> 1;            // 0..3 (M)
    const int wn   = wv & 1;             // 0..1 (N)
    const int hf   = (lane >> 4) & 1;
    const int lr   = lane & 15;
    const int NSTEP = EMBED / 32;        // 32

    if (wv == 0) {
        tdm_load_2d(lds_off_u32(a_lds), Ag, 32, 128, EMBED);
        tdm_load_2d(lds_off_u32(b_lds), Bg, 32, 128, EMBED);
    }
    for (int j = 0; j < NSTEP; ++j) {
        __syncthreads();                 // all waves done reading buf[(j+1)&1]
        if (wv == 0) {
            int nk = (j + 1 < NSTEP) ? (j + 1) * 32 : 0;   // dummy re-issue on last
            int nb = (j + 1) & 1;
            tdm_load_2d(lds_off_u32(a_lds + nb * (128 * 32)), Ag + nk, 32, 128, EMBED);
            tdm_load_2d(lds_off_u32(b_lds + nb * (128 * 32)), Bg + nk, 32, 128, EMBED);
            __builtin_amdgcn_s_wait_tensorcnt(2);          // tile j has landed
        }
        __syncthreads();                 // tile j visible to all waves
        const unsigned short* at = a_lds + (j & 1) * (128 * 32);
        const unsigned short* bt = b_lds + (j & 1) * (128 * 32);

        v16bf afr[2];
        #pragma unroll
        for (int mi = 0; mi < 2; ++mi) {
            int r = wm * 32 + mi * 16 + lr;
            const unsigned short* p = at + r * 32 + hf * 8;    // A-layout
            afr[mi] = load_frag(p, p + 16);
        }
        #pragma unroll
        for (int ni = 0; ni < 4; ++ni) {
            int nr = wn * 64 + ni * 16 + lr;
            const unsigned short* p = bt + nr * 32 + hf * 8;   // B-layout
            v16bf bfr = load_frag(p, p + 16);
            #pragma unroll
            for (int mi = 0; mi < 2; ++mi)
                acc[mi][ni] = wmma_bf16(afr[mi], bfr, acc[mi][ni]);
        }
    }
}

// ---------------- kernel 1: QKV projection + head scatter ----------------
__global__ void __launch_bounds__(256) qkv_gemm(
    const unsigned short* __restrict__ xb,     // [8192][1024]
    const unsigned short* __restrict__ wt,     // [3072][1024]  (W_qkv^T)
    const float* __restrict__ bias,            // [3072]
    unsigned short* __restrict__ Qw,           // [B*H][T][64]  (pre-scaled 1/8)
    unsigned short* __restrict__ Kw,           // [B*H][T][64]
    unsigned short* __restrict__ Vt)           // [B*H][64][T]  (transposed)
{
    __shared__ __align__(16) unsigned short a_lds[2][128 * 32];
    __shared__ __align__(16) unsigned short b_lds[2][128 * 32];
    const int m0 = blockIdx.x * 128;
    const int n0 = blockIdx.y * 128;

    v8f zero = {0.f, 0.f, 0.f, 0.f, 0.f, 0.f, 0.f, 0.f};
    v8f acc[2][4];
    #pragma unroll
    for (int i = 0; i < 2; ++i)
        #pragma unroll
        for (int j = 0; j < 4; ++j) acc[i][j] = zero;

    gemm_core_128x128(xb + (size_t)m0 * EMBED, wt + (size_t)n0 * EMBED,
                      &a_lds[0][0], &b_lds[0][0], acc);

    const int lane = threadIdx.x & 31;
    const int wv = threadIdx.x >> 5;
    const int wm = wv >> 1, wn = wv & 1;
    const int hf = (lane >> 4) & 1, lr = lane & 15;
    const int which = n0 >> 10;   // scalar: whole block lies in one Q/K/V section

    if (which == 0) {
        #pragma unroll
        for (int mi = 0; mi < 2; ++mi)
            #pragma unroll
            for (int ni = 0; ni < 4; ++ni) {
                int n = n0 + wn * 64 + ni * 16 + lr;
                float bv = bias[n];
                int c = n & 1023, h = c >> 6, d = c & 63;
                #pragma unroll
                for (int v = 0; v < 8; ++v) {
                    int m = m0 + wm * 32 + mi * 16 + v + 8 * hf;
                    int bh = (m >> 11) * HEADS + h;
                    int t = m & 2047;
                    Qw[((size_t)bh * SEQ + t) * HDIM + d] =
                        f32_to_bf16((acc[mi][ni][v] + bv) * 0.125f);
                }
            }
    } else if (which == 1) {
        #pragma unroll
        for (int mi = 0; mi < 2; ++mi)
            #pragma unroll
            for (int ni = 0; ni < 4; ++ni) {
                int n = n0 + wn * 64 + ni * 16 + lr;
                float bv = bias[n];
                int c = n & 1023, h = c >> 6, d = c & 63;
                #pragma unroll
                for (int v = 0; v < 8; ++v) {
                    int m = m0 + wm * 32 + mi * 16 + v + 8 * hf;
                    int bh = (m >> 11) * HEADS + h;
                    int t = m & 2047;
                    Kw[((size_t)bh * SEQ + t) * HDIM + d] =
                        f32_to_bf16(acc[mi][ni][v] + bv);
                }
            }
    } else {
        #pragma unroll
        for (int mi = 0; mi < 2; ++mi)
            #pragma unroll
            for (int ni = 0; ni < 4; ++ni) {
                int n = n0 + wn * 64 + ni * 16 + lr;
                float bv = bias[n];
                int c = n & 1023, h = c >> 6, d = c & 63;
                #pragma unroll
                for (int v = 0; v < 8; ++v) {
                    int m = m0 + wm * 32 + mi * 16 + v + 8 * hf;
                    int bh = (m >> 11) * HEADS + h;
                    int t = m & 2047;
                    Vt[((size_t)bh * HDIM + d) * SEQ + t] =
                        f32_to_bf16(acc[mi][ni][v] + bv);
                }
            }
    }
}

// ---------------- kernel 2: flash attention (causal), TDM K/V staging ----------------
__global__ void __launch_bounds__(256) flash_attn(
    const unsigned short* __restrict__ Qw,
    const unsigned short* __restrict__ Kw,
    const unsigned short* __restrict__ Vt,
    unsigned short* __restrict__ Y)            // [B][T][C] bf16
{
    __shared__ __align__(16) unsigned short k_lds[2][64 * 64];     // [key][d]
    __shared__ __align__(16) unsigned short v_lds[2][64 * 64];     // [d][key]
    __shared__ __align__(16) unsigned short p_lds[8][16 * 64];     // per-wave P tile

    const int qt = blockIdx.x & 15;
    const int h  = (blockIdx.x >> 4) & 15;
    const int b  = blockIdx.x >> 8;
    const int bh = b * HEADS + h;
    const int q0 = qt * 128;

    const int tid  = threadIdx.x;
    const int lane = tid & 31;
    const int wv   = tid >> 5;
    const int hf   = (lane >> 4) & 1;
    const int lr   = lane & 15;
    const int qrow0 = q0 + wv * 16;

    // Q fragments in registers (D=64 -> 2 K-steps)
    const unsigned short* qbase = Qw + ((size_t)bh * SEQ + qrow0 + lr) * HDIM;
    v16bf qf[2];
    #pragma unroll
    for (int ks0 = 0; ks0 < 2; ++ks0) {
        const unsigned short* p = qbase + ks0 * 32 + hf * 8;
        qf[ks0] = load_frag(p, p + 16);
    }

    v8f zero = {0.f, 0.f, 0.f, 0.f, 0.f, 0.f, 0.f, 0.f};
    v8f o[4];
    #pragma unroll
    for (int t = 0; t < 4; ++t) o[t] = zero;
    float rmax[8], rsum[8];
    #pragma unroll
    for (int v = 0; v < 8; ++v) { rmax[v] = -3.0e38f; rsum[v] = 0.f; }

    const unsigned short* Kbase = Kw + (size_t)bh * SEQ * HDIM;
    const unsigned short* Vbase = Vt + (size_t)bh * HDIM * SEQ;
    const int nkb = (q0 >> 6) + 2;

    if (wv == 0) {
        tdm_load_2d(lds_off_u32(&k_lds[0][0]), Kbase, HDIM, 64, HDIM);
        tdm_load_2d(lds_off_u32(&v_lds[0][0]), Vbase, 64, HDIM, SEQ);
    }
    for (int j = 0; j < nkb; ++j) {
        const int kb = j * 64;
        __syncthreads();                          // done reading buf[(j+1)&1]
        if (wv == 0) {
            int nkb2 = (j + 1 < nkb) ? (j + 1) * 64 : 0;   // dummy on last iter
            int nb = (j + 1) & 1;
            tdm_load_2d(lds_off_u32(&k_lds[nb][0]), Kbase + (size_t)nkb2 * HDIM,
                        HDIM, 64, HDIM);
            tdm_load_2d(lds_off_u32(&v_lds[nb][0]), Vbase + nkb2, 64, HDIM, SEQ);
            __builtin_amdgcn_s_wait_tensorcnt(2); // tile j complete
        }
        __syncthreads();                          // tile j visible
        const unsigned short* kt = &k_lds[j & 1][0];
        const unsigned short* vt = &v_lds[j & 1][0];

        // S = Q * K^T (1/sqrt(D) folded into Q)
        v8f s[4];
        #pragma unroll
        for (int t = 0; t < 4; ++t) s[t] = zero;
        #pragma unroll
        for (int t = 0; t < 4; ++t) {
            const unsigned short* kp = kt + (t * 16 + lr) * 64;
            #pragma unroll
            for (int ks2 = 0; ks2 < 2; ++ks2) {
                const unsigned short* p = kp + ks2 * 32 + hf * 8;
                s[t] = wmma_bf16(qf[ks2], load_frag(p, p + 16), s[t]);
            }
        }

        // causal mask: C/D layout row = v + 8*hf, col = lane&15
        #pragma unroll
        for (int t = 0; t < 4; ++t) {
            int key = kb + t * 16 + lr;
            #pragma unroll
            for (int v = 0; v < 8; ++v) {
                int qrow = qrow0 + v + 8 * hf;
                s[t][v] = (key > qrow) ? -3.0e38f : s[t][v];
            }
        }

        // online softmax (row reductions across 16 lanes of each half)
        #pragma unroll
        for (int v = 0; v < 8; ++v) {
            float mx = fmaxf(fmaxf(s[0][v], s[1][v]), fmaxf(s[2][v], s[3][v]));
            mx = fmaxf(mx, __shfl_xor(mx, 1, 32));
            mx = fmaxf(mx, __shfl_xor(mx, 2, 32));
            mx = fmaxf(mx, __shfl_xor(mx, 4, 32));
            mx = fmaxf(mx, __shfl_xor(mx, 8, 32));
            float nm = fmaxf(rmax[v], mx);
            float alpha = __expf(rmax[v] - nm);
            rmax[v] = nm;
            rsum[v] *= alpha;
            #pragma unroll
            for (int t = 0; t < 4; ++t) o[t][v] *= alpha;
            float ps = 0.f;
            #pragma unroll
            for (int t = 0; t < 4; ++t) {
                float p = __expf(s[t][v] - nm);
                s[t][v] = p;
                ps += p;
            }
            ps += __shfl_xor(ps, 1, 32);
            ps += __shfl_xor(ps, 2, 32);
            ps += __shfl_xor(ps, 4, 32);
            ps += __shfl_xor(ps, 8, 32);
            rsum[v] += ps;
        }

        // C/D-layout f32 -> A-layout bf16 via wave-local LDS (DS in-order per wave)
        unsigned short* pw = &p_lds[wv][0];
        #pragma unroll
        for (int t = 0; t < 4; ++t)
            #pragma unroll
            for (int v = 0; v < 8; ++v)
                pw[(v + 8 * hf) * 64 + t * 16 + lr] = f32_to_bf16(s[t][v]);

        // O += P * V  (B operand = V^T tile, K-contiguous)
        #pragma unroll
        for (int t = 0; t < 4; ++t) {
            const unsigned short* vp = vt + (t * 16 + lr) * 64;
            #pragma unroll
            for (int ks2 = 0; ks2 < 2; ++ks2) {
                const unsigned short* ap = pw + lr * 64 + ks2 * 32 + hf * 8;
                v16bf a = load_frag(ap, ap + 16);
                const unsigned short* bp = vp + ks2 * 32 + hf * 8;
                o[t] = wmma_bf16(a, load_frag(bp, bp + 16), o[t]);
            }
        }
    }

    // normalize + write y[b, t, h*64+d] as bf16
    #pragma unroll
    for (int v = 0; v < 8; ++v) rsum[v] = 1.0f / rsum[v];
    #pragma unroll
    for (int t = 0; t < 4; ++t) {
        int col = h * HDIM + t * 16 + lr;
        #pragma unroll
        for (int v = 0; v < 8; ++v) {
            int row = qrow0 + v + 8 * hf;
            Y[((size_t)b * SEQ + row) * EMBED + col] = f32_to_bf16(o[t][v] * rsum[v]);
        }
    }
}

// ---------------- kernel 3: output projection ----------------
__global__ void __launch_bounds__(256) proj_gemm(
    const unsigned short* __restrict__ yb,     // [8192][1024] bf16
    const unsigned short* __restrict__ wt,     // [1024][1024] (W_proj^T)
    const float* __restrict__ bias,            // [1024]
    float* __restrict__ out)                   // [8192][1024] f32
{
    __shared__ __align__(16) unsigned short a_lds[2][128 * 32];
    __shared__ __align__(16) unsigned short b_lds[2][128 * 32];
    const int m0 = blockIdx.x * 128;
    const int n0 = blockIdx.y * 128;

    v8f zero = {0.f, 0.f, 0.f, 0.f, 0.f, 0.f, 0.f, 0.f};
    v8f acc[2][4];
    #pragma unroll
    for (int i = 0; i < 2; ++i)
        #pragma unroll
        for (int j = 0; j < 4; ++j) acc[i][j] = zero;

    gemm_core_128x128(yb + (size_t)m0 * EMBED, wt + (size_t)n0 * EMBED,
                      &a_lds[0][0], &b_lds[0][0], acc);

    const int lane = threadIdx.x & 31;
    const int wv = threadIdx.x >> 5;
    const int wm = wv >> 1, wn = wv & 1;
    const int hf = (lane >> 4) & 1, lr = lane & 15;

    #pragma unroll
    for (int mi = 0; mi < 2; ++mi) {
        #pragma unroll
        for (int ni = 0; ni < 4; ++ni) {
            int n = n0 + wn * 64 + ni * 16 + lr;
            float bv = bias[n];
            #pragma unroll
            for (int v = 0; v < 8; ++v) {
                int m = m0 + wm * 32 + mi * 16 + v + 8 * hf;
                out[(size_t)m * EMBED + n] = acc[mi][ni][v] + bv;
            }
        }
    }
}

// ---------------- host launcher ----------------
extern "C" void kernel_launch(void* const* d_in, const int* in_sizes, int n_in,
                              void* d_out, int out_size, void* d_ws, size_t ws_size,
                              hipStream_t stream) {
    const float* x      = (const float*)d_in[0];
    const float* W_qkv  = (const float*)d_in[1];
    const float* b_qkv  = (const float*)d_in[2];
    const float* W_proj = (const float*)d_in[3];
    const float* b_proj = (const float*)d_in[4];
    float* out = (float*)d_out;

    char* ws = (char*)d_ws;
    size_t off = 0;
    unsigned short* xb     = (unsigned short*)(ws + off); off += (size_t)ROWS * EMBED * 2;
    unsigned short* wqkvT  = (unsigned short*)(ws + off); off += (size_t)QKV_N * EMBED * 2;
    unsigned short* wprojT = (unsigned short*)(ws + off); off += (size_t)EMBED * EMBED * 2;
    unsigned short* Qw     = (unsigned short*)(ws + off); off += (size_t)ROWS * EMBED * 2;
    unsigned short* Kw     = (unsigned short*)(ws + off); off += (size_t)ROWS * EMBED * 2;
    unsigned short* Vt     = (unsigned short*)(ws + off); off += (size_t)ROWS * EMBED * 2;
    unsigned short* yb     = (unsigned short*)(ws + off); off += (size_t)ROWS * EMBED * 2;
    (void)ws_size; (void)in_sizes; (void)n_in; (void)out_size;

    cvt_f32_to_bf16<<<4096, 256, 0, stream>>>(x, xb, ROWS * EMBED);
    transpose_to_bf16<<<4096, 256, 0, stream>>>(W_qkv, wqkvT, QKV_N);
    transpose_to_bf16<<<2048, 256, 0, stream>>>(W_proj, wprojT, EMBED);

    dim3 g1(ROWS / 128, QKV_N / 128);   // 64 x 24
    qkv_gemm<<<g1, 256, 0, stream>>>(xb, wqkvT, b_qkv, Qw, Kw, Vt);

    flash_attn<<<BATCH * HEADS * (SEQ / 128), 256, 0, stream>>>(Qw, Kw, Vt, yb);

    dim3 g2(ROWS / 128, EMBED / 128);   // 64 x 8
    proj_gemm<<<g2, 256, 0, stream>>>(yb, wprojT, b_proj, out);
}
